// LSTMDense_28020366639387
// MI455X (gfx1250) — matrix operand order repaired and checked by reading, not compiled, not run on hardware
//
#include <hip/hip_runtime.h>
#include <hip/hip_bf16.h>
#include <math.h>

typedef __attribute__((ext_vector_type(16))) __bf16 bf16x16;
typedef __attribute__((ext_vector_type(8)))  float  f32x8;
typedef unsigned int   u32;
typedef unsigned short u16;
typedef __attribute__((ext_vector_type(4))) u32 u32x4;   // native 16B vector

constexpr int kH   = 256;      // hidden
constexpr int kD   = 128;      // input feature
constexpr int kT   = 1024;     // seq len
constexpr int kB   = 256;      // batch
constexpr int kBT  = 16;       // batch tile per workgroup
constexpr int kNWG = kB / kBT; // 16 workgroups
constexpr int kG   = 4 * kH;   // 1024 gate rows
constexpr int kC   = 10;       // classes
constexpr int kThreads = 512;  // 16 wave32s

// LDS pitches in halves: row stride 272B/528B -> 16B aligned with a
// +4-dword skew per row => conflict-free ds_load_b128 A-fragments.
constexpr int kXP = kD + 8;    // 136
constexpr int kHP = kH + 8;    // 264

// ---------------- helpers ----------------

__device__ __forceinline__ u16 f2bf(float f) {
  union { float f; u32 u; } v; v.f = f;
  u32 u = v.u;
  u32 lsb = (u >> 16) & 1u;
  u += 0x7fffu + lsb;               // round-to-nearest-even
  return (u16)(u >> 16);
}

__device__ __forceinline__ float sigf(float x) {
  return 1.0f / (1.0f + __expf(-x));
}
__device__ __forceinline__ float tanh_fast(float x) {
  x = fminf(15.0f, fmaxf(-15.0f, x));
  float e = __expf(2.0f * x);
  return (e - 1.0f) / (e + 1.0f);
}

// Launder a pointer so LICM cannot prove loads through it loop-invariant:
// keeps the weight stream as in-loop NT global loads (re-read from L2 each
// timestep) instead of being hoisted out of the t-loop and spilled to scratch.
__device__ __forceinline__ const u16* opaque(const u16* p) {
  asm volatile("" : "+v"(p));
  return p;
}

// A-fragment (16x32 bf16, contiguous k = 16*h + e per lane half) from LDS.
__device__ __forceinline__ bf16x16 load_fragA(const u16* lds, int pitch, int m,
                                              int s, int kbase) {
  union { bf16x16 v; u32x4 q[2]; } u;
  const u16* p = lds + m * pitch + kbase + s * 16;
  u.q[0] = *(const u32x4*)(p);
  u.q[1] = *(const u32x4*)(p + 8);
  return u.v;
}

// B-fragment (32x16 bf16): two contiguous 16B non-temporal loads per lane.
__device__ __forceinline__ bf16x16 load_fragB(const u16* p) {
  union { bf16x16 v; u32x4 q[2]; } u;
  u.q[0] = __builtin_nontemporal_load((const u32x4*)p);
  u.q[1] = __builtin_nontemporal_load(((const u32x4*)p) + 1);
  return u.v;
}

#define WMMA_BF16(A, B, C)                                                   \
  __builtin_amdgcn_wmma_f32_16x16x32_bf16(false, (A), false, (B), (short)0,  \
                                          (C), false, false)

// One K-segment of the gate GEMM: acc[q] += A(16x32*NKT) x Bq(32*NKT x 16),
// software-pipelined B-fragment stream (ping-pong, last pair peeled).
// bq pointers already include this wave's tile base and the per-lane offset;
// consecutive kt blocks are 512 halves apart.
template <int NKT>
__device__ __forceinline__ void gemm_seg(f32x8 acc[4], const u16* aLds,
                                         int apitch, int m, int s,
                                         const u16* b0, const u16* b1,
                                         const u16* b2, const u16* b3) {
  __builtin_amdgcn_sched_barrier(0);
  b0 = opaque(b0); b1 = opaque(b1); b2 = opaque(b2); b3 = opaque(b3);
  bf16x16 be[4], bo[4];
  be[0] = load_fragB(b0); be[1] = load_fragB(b1);
  be[2] = load_fragB(b2); be[3] = load_fragB(b3);
  #pragma unroll 1
  for (int kt = 0; kt < NKT - 2; kt += 2) {
    bo[0] = load_fragB(b0 + 512); bo[1] = load_fragB(b1 + 512);
    bo[2] = load_fragB(b2 + 512); bo[3] = load_fragB(b3 + 512);
    bf16x16 a = load_fragA(aLds, apitch, m, s, kt * 32);
    acc[0] = WMMA_BF16(a, be[0], acc[0]);
    acc[1] = WMMA_BF16(a, be[1], acc[1]);
    acc[2] = WMMA_BF16(a, be[2], acc[2]);
    acc[3] = WMMA_BF16(a, be[3], acc[3]);
    b0 += 1024; b1 += 1024; b2 += 1024; b3 += 1024;
    be[0] = load_fragB(b0); be[1] = load_fragB(b1);
    be[2] = load_fragB(b2); be[3] = load_fragB(b3);
    a = load_fragA(aLds, apitch, m, s, kt * 32 + 32);
    acc[0] = WMMA_BF16(a, bo[0], acc[0]);
    acc[1] = WMMA_BF16(a, bo[1], acc[1]);
    acc[2] = WMMA_BF16(a, bo[2], acc[2]);
    acc[3] = WMMA_BF16(a, bo[3], acc[3]);
  }
  // epilogue: kt = NKT-2, NKT-1 (be already holds NKT-2)
  bo[0] = load_fragB(b0 + 512); bo[1] = load_fragB(b1 + 512);
  bo[2] = load_fragB(b2 + 512); bo[3] = load_fragB(b3 + 512);
  bf16x16 a = load_fragA(aLds, apitch, m, s, (NKT - 2) * 32);
  acc[0] = WMMA_BF16(a, be[0], acc[0]);
  acc[1] = WMMA_BF16(a, be[1], acc[1]);
  acc[2] = WMMA_BF16(a, be[2], acc[2]);
  acc[3] = WMMA_BF16(a, be[3], acc[3]);
  a = load_fragA(aLds, apitch, m, s, (NKT - 1) * 32);
  acc[0] = WMMA_BF16(a, bo[0], acc[0]);
  acc[1] = WMMA_BF16(a, bo[1], acc[1]);
  acc[2] = WMMA_BF16(a, bo[2], acc[2]);
  acc[3] = WMMA_BF16(a, bo[3], acc[3]);
  __builtin_amdgcn_sched_barrier(0);
}

// ---------------- weight pre-conversion (f32 -> bf16, WMMA-B tiled) ---------
// dst block[(tile_r*nKt + kt)] of 512 halves laid out [s][r16][k16].

__global__ void cvt_weight_tiled(const float* __restrict__ src,
                                 u16* __restrict__ dst, int K, int nKt,
                                 int ktshift, int total) {
  int d = blockIdx.x * 256 + threadIdx.x;
  if (d >= total) return;
  int kin = d & 15;
  int r16 = (d >> 4) & 15;
  int s   = (d >> 8) & 1;
  int tmp = d >> 9;
  int kt  = tmp & (nKt - 1);
  int tr  = tmp >> ktshift;
  dst[d] = f2bf(src[(tr * 16 + r16) * K + (kt * 32 + s * 16 + kin)]);
}

// ---------------- fused 2-layer LSTM + dense + softmax ----------------
// One workgroup per 16 batch rows; wave w owns h-cols [16w,16w+16) and all
// four gates for them. Cell state lives in registers (C/D fragment layout);
// h lives in LDS as bf16; weights stream NT from L2 in fragment order.

__launch_bounds__(kThreads, 1)
__global__ void lstm_fused(const float* __restrict__ x,
                           const u16* __restrict__ Wih0t,
                           const u16* __restrict__ Whh0t,
                           const u16* __restrict__ Wih1t,
                           const u16* __restrict__ Whh1t,
                           const float* __restrict__ b0,
                           const float* __restrict__ b1,
                           const float* __restrict__ Wd,
                           const float* __restrict__ bd,
                           float* __restrict__ out) {
  __shared__ u16   xbuf[16 * kXP];
  __shared__ u16   h0buf[16 * kHP];
  __shared__ u16   h1buf[16 * kHP];
  __shared__ float hf[16 * kH];
  __shared__ float lg[kBT * kC];

  const int tid  = threadIdx.x;
  const int w    = tid >> 5;             // wave 0..15
  const int lane = tid & 31;
  const int s    = lane >> 4;
  const int nidx = lane & 15;
  const int col0 = w * 16;               // this wave's h-col base
  const int bb   = blockIdx.x * kBT;
  const int laneoff = (s * 16 + nidx) * 16;   // B-fragment per-lane offset

  for (int i = tid; i < 16 * kHP; i += kThreads) { h0buf[i] = 0; h1buf[i] = 0; }

  // Per-(matrix, gate) fragment base pointers and bias scalars: loop-invariant.
  const u16 *pI0[4], *pH0[4], *pI1[4], *pH1[4];
  float bv0[4], bv1[4];
  #pragma unroll
  for (int q = 0; q < 4; ++q) {
    int tr = q * 16 + w;
    pI0[q] = Wih0t + (tr << 11) + laneoff;   // nKt = 4 blocks of 512 halves
    pH0[q] = Whh0t + (tr << 12) + laneoff;   // nKt = 8
    pI1[q] = Wih1t + (tr << 12) + laneoff;
    pH1[q] = Whh1t + (tr << 12) + laneoff;
    bv0[q] = b0[q * kH + col0 + nidx];
    bv1[q] = b1[q * kH + col0 + nidx];
  }

  f32x8 c0, c1;
  #pragma unroll
  for (int v = 0; v < 8; ++v) { c0[v] = 0.0f; c1[v] = 0.0f; }

  __syncthreads();

  f32x8 acc[4];

  for (int t = 0; t < kT; ++t) {
    // ---- stage x_t (16 x 128 f32 -> bf16 LDS); prefetch x_{t+4} ----
    {
      int r  = tid >> 5;                 // 0..15
      int cb = (tid & 31) * 4;           // 0..124
      const float* ps = x + ((size_t)(bb + r) * kT + t) * kD + cb;
      float4 a0 = *(const float4*)(ps);
      if (t + 4 < kT) __builtin_prefetch(ps + 4 * kD, 0, 1);
      u16* pd = xbuf + r * kXP + cb;
      pd[0] = f2bf(a0.x); pd[1] = f2bf(a0.y);
      pd[2] = f2bf(a0.z); pd[3] = f2bf(a0.w);
    }
    __syncthreads();

    // ---- layer 0: g = x_t@Wih0^T + h0@Whh0^T + b0 ----
    #pragma unroll
    for (int q = 0; q < 4; ++q)
      #pragma unroll
      for (int v = 0; v < 8; ++v) acc[q][v] = bv0[q];
    gemm_seg<4>(acc, xbuf,  kXP, nidx, s, pI0[0], pI0[1], pI0[2], pI0[3]);
    gemm_seg<8>(acc, h0buf, kHP, nidx, s, pH0[0], pH0[1], pH0[2], pH0[3]);
    __syncthreads();                     // all waves done reading h0buf
    #pragma unroll
    for (int v = 0; v < 8; ++v) {
      float iv = sigf(acc[0][v]);
      float fv = sigf(acc[1][v]);
      float gv = tanh_fast(acc[2][v]);
      float ov = sigf(acc[3][v]);
      float cc = fv * c0[v] + iv * gv;
      c0[v] = cc;
      h0buf[(8 * s + v) * kHP + col0 + nidx] = f2bf(ov * tanh_fast(cc));
    }
    __syncthreads();                     // h0_new visible

    // ---- layer 1: g = h0@Wih1^T + h1@Whh1^T + b1 ----
    #pragma unroll
    for (int q = 0; q < 4; ++q)
      #pragma unroll
      for (int v = 0; v < 8; ++v) acc[q][v] = bv1[q];
    gemm_seg<8>(acc, h0buf, kHP, nidx, s, pI1[0], pI1[1], pI1[2], pI1[3]);
    gemm_seg<8>(acc, h1buf, kHP, nidx, s, pH1[0], pH1[1], pH1[2], pH1[3]);
    __syncthreads();                     // all waves done reading h1buf
    bool last = (t == kT - 1);
    #pragma unroll
    for (int v = 0; v < 8; ++v) {
      float iv = sigf(acc[0][v]);
      float fv = sigf(acc[1][v]);
      float gv = tanh_fast(acc[2][v]);
      float ov = sigf(acc[3][v]);
      float cc = fv * c1[v] + iv * gv;
      c1[v] = cc;
      float hv = ov * tanh_fast(cc);
      h1buf[(8 * s + v) * kHP + col0 + nidx] = f2bf(hv);
      if (last) hf[(8 * s + v) * kH + col0 + nidx] = hv;
    }
    __syncthreads();                     // h1_new visible; xbuf reusable
  }

  // ---- dense head + softmax (f32) ----
  if (tid < kBT * kC) {
    int r = tid / kC, cls = tid % kC;
    const float* wr = Wd + cls * kH;
    float a = bd[cls];
    for (int k = 0; k < kH; ++k) a += hf[r * kH + k] * wr[k];
    lg[tid] = a;
  }
  __syncthreads();
  if (tid < kBT) {
    float m = -1e30f;
    #pragma unroll
    for (int c = 0; c < kC; ++c) m = fmaxf(m, lg[tid * kC + c]);
    float e[kC], sum = 0.0f;
    #pragma unroll
    for (int c = 0; c < kC; ++c) { e[c] = __expf(lg[tid * kC + c] - m); sum += e[c]; }
    float inv = 1.0f / sum;
    #pragma unroll
    for (int c = 0; c < kC; ++c) out[(bb + tid) * kC + c] = e[c] * inv;
  }
}

// ---------------- host side ----------------

extern "C" void kernel_launch(void* const* d_in, const int* in_sizes, int n_in,
                              void* d_out, int out_size, void* d_ws, size_t ws_size,
                              hipStream_t stream) {
  const float* x    = (const float*)d_in[0];
  const float* Wih0 = (const float*)d_in[1];
  const float* Whh0 = (const float*)d_in[2];
  const float* b0   = (const float*)d_in[3];
  const float* Wih1 = (const float*)d_in[4];
  const float* Whh1 = (const float*)d_in[5];
  const float* b1   = (const float*)d_in[6];
  const float* Wd   = (const float*)d_in[7];
  const float* bd   = (const float*)d_in[8];

  u16* ws    = (u16*)d_ws;
  u16* Wih0t = ws;                       // 1024*128 halves
  u16* Whh0t = Wih0t + kG * kD;          // 1024*256 halves each below
  u16* Wih1t = Whh0t + kG * kH;
  u16* Whh1t = Wih1t + kG * kH;

  int tIh0 = kG * kD, tHh = kG * kH;
  cvt_weight_tiled<<<(tIh0 + 255) / 256, 256, 0, stream>>>(Wih0, Wih0t, kD, 4, 2, tIh0);
  cvt_weight_tiled<<<(tHh  + 255) / 256, 256, 0, stream>>>(Whh0, Whh0t, kH, 8, 3, tHh);
  cvt_weight_tiled<<<(tHh  + 255) / 256, 256, 0, stream>>>(Wih1, Wih1t, kH, 8, 3, tHh);
  cvt_weight_tiled<<<(tHh  + 255) / 256, 256, 0, stream>>>(Whh1, Whh1t, kH, 8, 3, tHh);

  lstm_fused<<<kNWG, kThreads, 0, stream>>>(x, Wih0t, Whh0t, Wih1t, Whh1t,
                                            b0, b1, Wd, bd, (float*)d_out);
}
